// GNN_36764920054007
// MI455X (gfx1250) — matrix operand order repaired and checked by reading, not compiled, not run on hardware
//
#include <hip/hip_runtime.h>
#include <hip/hip_bf16.h>
#include <math.h>

// ---------------- problem constants ----------------
#define NN      50000
#define EE      1600000
#define F_IN    128
#define HH      256
#define ED      64
#define NL      8
#define NC      10
#define NG      64
#define NEG_SLOPE 0.2f

// ---------------- WMMA helpers (gfx1250, wave32) ----------------
typedef __attribute__((ext_vector_type(16))) _Float16 v16h;
typedef __attribute__((ext_vector_type(8)))  _Float16 v8h;
typedef __attribute__((ext_vector_type(4)))  _Float16 v4h;
typedef __attribute__((ext_vector_type(8)))  float    v8f;

__device__ inline v8f wmma16(v16h a, v16h b, v8f c) {
  // D = A(16x32 f16) * B(32x16 f16) + C(16x16 f32)
  return __builtin_amdgcn_wmma_f32_16x16x32_f16(false, a, false, b, (short)0, c,
                                                false, false);
}

// A-matrix 16x32 f16 from row-major source (leading dim lda halves).
// ISA layout: lane L holds row M=L%16; halves 0..7 = K {base..base+7},
// halves 8..15 = K {base+16..base+23}, base = 8*(L/16).
// Rows must be 16B aligned (lda multiple of 8) -> two b128 loads.
__device__ inline v16h load_a16(const _Float16* __restrict__ A, int lda) {
  const int lane = threadIdx.x & 31;
  const _Float16* p = A + (lane & 15) * lda + ((lane >> 4) << 3);
  v8h lo = *(const v8h*)(p);
  v8h hi = *(const v8h*)(p + 16);
  v16h a;
#pragma unroll
  for (int i = 0; i < 8; ++i) { a[i] = lo[i]; a[8 + i] = hi[i]; }
  return a;
}

// B fragments are pre-swizzled into lane-major tiles: tile t occupies
// P[t*512 .. t*512+511], lane L's 16 halves contiguous at t*512 + L*16.
// -> one 32-byte contiguous load per lane (two global_load_b128).
__device__ inline v16h load_bp(const _Float16* __restrict__ P, int tile) {
  const int lane = threadIdx.x & 31;
  return *(const v16h*)(P + ((size_t)tile << 9) + (lane << 4));
}

// D/C f32 layout: element (M = r + 8*(L/16), N = L%16) lives in c[r].

__device__ inline void atomicMaxF(float* addr, float val) {
  if (val >= 0.0f) atomicMax((int*)addr, __float_as_int(val));
  else             atomicMin((unsigned int*)addr, __float_as_uint(val));
}

// ---------------- weight repack: fp32 (K x N) -> f16 WMMA-B tiles ----------
// Tile index = nt * (K/32) + kt ; within tile: lane = idx/16, j = idx%16,
// source (k, n) = (kt*32 + (j>>3)*16 + (lane>>4)*8 + (j&7), nt*16 + (lane&15)).
__global__ void k_pack_b(const float* __restrict__ W, _Float16* __restrict__ P,
                         int K, int N) {
  const int KT = K >> 5;
  long long i = (long long)blockIdx.x * blockDim.x + threadIdx.x;
  long long st = (long long)gridDim.x * blockDim.x;
  long long tot = (long long)K * N;
  for (; i < tot; i += st) {
    long long tile = i >> 9;
    int r = (int)(i & 511);
    int lane = r >> 4, j = r & 15;
    int nt = (int)(tile / KT), kt = (int)(tile % KT);
    int k = kt * 32 + ((j >> 3) << 4) + ((lane >> 4) << 3) + (j & 7);
    int n = nt * 16 + (lane & 15);
    P[i] = (_Float16)W[(size_t)k * N + n];
  }
}

// ---------------- small utility kernels ----------------
__global__ void k_fill(float* __restrict__ p, float v, long long n) {
  long long i = (long long)blockIdx.x * blockDim.x + threadIdx.x;
  long long st = (long long)gridDim.x * blockDim.x;
  for (; i < n; i += st) p[i] = v;
}

__global__ void k_rows_bias(float* __restrict__ p, const float* __restrict__ bias,
                            long long rows, int cols) {
  long long i = (long long)blockIdx.x * blockDim.x + threadIdx.x;
  long long st = (long long)gridDim.x * blockDim.x;
  long long tot = rows * cols;
  for (; i < tot; i += st) p[i] = bias[i % cols];
}

// ---------------- K0: permutation-invariant edge MLP ----------------
__global__ __launch_bounds__(32) void k_edge_perm_net(
    const float* __restrict__ raw,       // E x 8
    const float* __restrict__ w1,        // 8 x 256
    const float* __restrict__ b1,        // 256
    const _Float16* __restrict__ w2p,    // packed 256x64 (KT=8, NT=4)
    const float* __restrict__ b2,        // 64
    const _Float16* __restrict__ wsp,    // packed 64x64 (KT=2, NT=4)
    const float* __restrict__ ws_full,   // 67 x 64 fp32 (rows 64..66 used)
    const float* __restrict__ bs,        // 64
    _Float16* __restrict__ ea16,         // E x 64 out
    int Ecnt) {
  __shared__ float xs_s[16][NL];
  __shared__ float st_s[16][3];                       // range, std, max
  __shared__ alignas(32) _Float16 h_s[16][HH];
  __shared__ alignas(32) _Float16 f_s[16][ED];
  const int lane = threadIdx.x;
  const int e0 = blockIdx.x * 16;

  if (lane < 16) {
    int e = min(e0 + lane, Ecnt - 1);
    float v[8];
#pragma unroll
    for (int i = 0; i < 8; ++i) v[i] = raw[(size_t)e * 8 + i];
#define CS(i, j) { float _a = v[i], _b = v[j]; v[i] = fminf(_a, _b); v[j] = fmaxf(_a, _b); }
    CS(0,1) CS(2,3) CS(4,5) CS(6,7)
    CS(0,2) CS(1,3) CS(4,6) CS(5,7)
    CS(1,2) CS(5,6) CS(0,4) CS(3,7)
    CS(1,5) CS(2,6)
    CS(1,4) CS(3,6)
    CS(2,4) CS(3,5)
    CS(3,4)
#undef CS
    float mean = 0.f;
#pragma unroll
    for (int i = 0; i < 8; ++i) { xs_s[lane][i] = v[i]; mean += v[i]; }
    mean *= 0.125f;
    float var = 0.f;
#pragma unroll
    for (int i = 0; i < 8; ++i) { float d = v[i] - mean; var += d * d; }
    var *= (1.0f / 7.0f);                              // ddof = 1
    st_s[lane][0] = v[7] - v[0];
    st_s[lane][1] = sqrtf(var);
    st_s[lane][2] = v[7];
  }
  __syncthreads();

  // hidden = relu(xs @ w1 + b1)  (K=8, scalar)
  for (int e = 0; e < 16; ++e) {
    for (int c = lane; c < HH; c += 32) {
      float acc = b1[c];
#pragma unroll
      for (int k = 0; k < 8; ++k) acc += xs_s[e][k] * w1[k * HH + c];
      h_s[e][c] = (_Float16)fmaxf(acc, 0.f);
    }
  }
  __syncthreads();

  // f = hidden @ w2 + b2   (16x256 @ 256x64)
  v16h a[8];
#pragma unroll
  for (int k = 0; k < 8; ++k) a[k] = load_a16(&h_s[0][0] + k * 32, HH);
  const int col0 = lane & 15, rh = (lane >> 4) << 3;
#pragma unroll
  for (int nt = 0; nt < 4; ++nt) {
    v8f acc = {};
#pragma unroll
    for (int k = 0; k < 8; ++k) acc = wmma16(a[k], load_bp(w2p, nt * 8 + k), acc);
    int col = nt * 16 + col0;
    float bv = b2[col];
#pragma unroll
    for (int r = 0; r < 8; ++r) f_s[r + rh][col] = (_Float16)(acc[r] + bv);
  }
  __syncthreads();

  // out = relu(f @ ws[0:64] + range*ws[64] + std*ws[65] + max*ws[66] + bs)
  v16h fa0 = load_a16(&f_s[0][0], ED);
  v16h fa1 = load_a16(&f_s[0][0] + 32, ED);
#pragma unroll
  for (int nt = 0; nt < 4; ++nt) {
    v8f acc = {};
    acc = wmma16(fa0, load_bp(wsp, nt * 2 + 0), acc);
    acc = wmma16(fa1, load_bp(wsp, nt * 2 + 1), acc);
    int col = nt * 16 + col0;
    float wr  = ws_full[64 * ED + col];
    float wsd = ws_full[65 * ED + col];
    float wmx = ws_full[66 * ED + col];
    float bv  = bs[col];
#pragma unroll
    for (int r = 0; r < 8; ++r) {
      int m = r + rh;
      float o = acc[r] + st_s[m][0] * wr + st_s[m][1] * wsd + st_s[m][2] * wmx + bv;
      int e = e0 + m;
      if (e < Ecnt) ea16[(size_t)e * ED + col] = (_Float16)fmaxf(o, 0.f);
    }
  }
}

// ---------------- node linears: xl = act(X)@WL + bl, xr = act(X)@WR ----------
template <int NK>  // NK = Kdim/32
__global__ __launch_bounds__(32) void k_node_linear(
    const float* __restrict__ X, int Kdim, int do_relu,
    const _Float16* __restrict__ WLP, const float* __restrict__ BL,
    const _Float16* __restrict__ WRP,
    float* __restrict__ XL, float* __restrict__ XR, int Nn) {
  __shared__ alignas(32) _Float16 a_s[16][HH];
  const int lane = threadIdx.x;
  const int n0 = blockIdx.x * 16;
  const int q = Kdim >> 2;
  for (int r = 0; r < 16; ++r) {
    int n = min(n0 + r, Nn - 1);
    const float4* px = (const float4*)(X + (size_t)n * Kdim);
    v4h* pa = (v4h*)&a_s[r][0];
    for (int c = lane; c < q; c += 32) {
      float4 xv = px[c];
      if (do_relu) {
        xv.x = fmaxf(xv.x, 0.f); xv.y = fmaxf(xv.y, 0.f);
        xv.z = fmaxf(xv.z, 0.f); xv.w = fmaxf(xv.w, 0.f);
      }
      v4h hv = { (_Float16)xv.x, (_Float16)xv.y, (_Float16)xv.z, (_Float16)xv.w };
      pa[c] = hv;
    }
  }
  __syncthreads();
  v16h a[NK];
#pragma unroll
  for (int k = 0; k < NK; ++k) a[k] = load_a16(&a_s[0][0] + k * 32, HH);
  const int col0 = lane & 15, rh = (lane >> 4) << 3;
  for (int nt = 0; nt < 16; ++nt) {
    v8f aL = {}, aR = {};
#pragma unroll
    for (int k = 0; k < NK; ++k) {
      aL = wmma16(a[k], load_bp(WLP, nt * NK + k), aL);
      aR = wmma16(a[k], load_bp(WRP, nt * NK + k), aR);
    }
    int col = nt * 16 + col0;
    float bv = BL[col];
#pragma unroll
    for (int r = 0; r < 8; ++r) {
      int n = n0 + r + rh;
      if (n < Nn) {
        XL[(size_t)n * HH + col] = aL[r] + bv;
        XR[(size_t)n * HH + col] = aR[r];
      }
    }
  }
}

// ---------------- GAT attention logits (fused e-proj + gather + att dot) ----
__global__ __launch_bounds__(32) void k_gat_logits(
    const _Float16* __restrict__ EA16,  // E x 64
    const _Float16* __restrict__ WEP,   // packed 64x256 (KT=2, NT=16)
    const float* __restrict__ XL, const float* __restrict__ XR,  // N x 256
    const float* __restrict__ ATT,                               // 256
    const int* __restrict__ SRC, const int* __restrict__ DST,
    float* __restrict__ LOGIT, float* __restrict__ M, int Ecnt) {
  __shared__ alignas(16) float xls[16][HH];
  __shared__ alignas(16) float xrs[16][HH];
  __shared__ int idx_s[16][2];
  const int lane = threadIdx.x;
  const int e0 = blockIdx.x * 16;
  if (lane < 16) {
    int e = min(e0 + lane, Ecnt - 1);
    idx_s[lane][0] = SRC[e];
    idx_s[lane][1] = DST[e];
  }
  __syncthreads();
  for (int r = 0; r < 16; ++r) {
    const float4* ps = (const float4*)(XL + (size_t)idx_s[r][0] * HH);
    const float4* pd = (const float4*)(XR + (size_t)idx_s[r][1] * HH);
    float4* ql = (float4*)&xls[r][0];
    float4* qr = (float4*)&xrs[r][0];
    for (int c = lane; c < HH / 4; c += 32) { ql[c] = ps[c]; qr[c] = pd[c]; }
  }
  __syncthreads();
  v16h a0 = load_a16(EA16 + (size_t)e0 * ED, ED);
  v16h a1 = load_a16(EA16 + (size_t)e0 * ED + 32, ED);
  float acc8[8];
#pragma unroll
  for (int r = 0; r < 8; ++r) acc8[r] = 0.f;
  const int col0 = lane & 15, rh = (lane >> 4) << 3;
  for (int nt = 0; nt < 16; ++nt) {
    v8f acc = {};
    acc = wmma16(a0, load_bp(WEP, nt * 2 + 0), acc);
    acc = wmma16(a1, load_bp(WEP, nt * 2 + 1), acc);
    int col = nt * 16 + col0;
    float attv = ATT[col];
#pragma unroll
    for (int r = 0; r < 8; ++r) {
      int m = r + rh;
      float s = acc[r] + xls[m][col] + xrs[m][col];
      s = (s > 0.f) ? s : NEG_SLOPE * s;
      acc8[r] += s * attv;
    }
  }
#pragma unroll
  for (int r = 0; r < 8; ++r) {  // reduce over the 16 lanes of each half-wave
    float v = acc8[r];
    v += __shfl_xor(v, 1, 32);
    v += __shfl_xor(v, 2, 32);
    v += __shfl_xor(v, 4, 32);
    v += __shfl_xor(v, 8, 32);
    acc8[r] = v;
  }
  if (col0 == 0) {
#pragma unroll
    for (int r = 0; r < 8; ++r) {
      int m = r + rh, e = e0 + m;
      if (e < Ecnt) {
        LOGIT[e] = acc8[r];
        atomicMaxF(&M[idx_s[m][1]], acc8[r]);
      }
    }
  }
}

// ---------------- softmax pieces ----------------
__global__ void k_expsum(const float* __restrict__ LOGIT, const float* __restrict__ M,
                         const int* __restrict__ DST, float* __restrict__ EX,
                         float* __restrict__ DEN, int Ecnt) {
  int e = blockIdx.x * blockDim.x + threadIdx.x;
  if (e >= Ecnt) return;
  int d = DST[e];
  float ex = __expf(LOGIT[e] - M[d]);
  EX[e] = ex;
  atomicAdd(&DEN[d], ex);
}

__global__ void k_scatter(const float* __restrict__ EX, const float* __restrict__ DEN,
                          const int* __restrict__ SRC, const int* __restrict__ DST,
                          const float* __restrict__ XL, float* __restrict__ OUT,
                          long long total) {
  long long i = (long long)blockIdx.x * blockDim.x + threadIdx.x;
  long long st = (long long)gridDim.x * blockDim.x;
  for (; i < total; i += st) {
    int e = (int)(i >> 8);
    int c = (int)(i & 255);
    int d = DST[e];
    float alpha = EX[e] / DEN[d];
    atomicAdd(&OUT[(size_t)d * HH + c], XL[(size_t)SRC[e] * HH + c] * alpha);
  }
}

// ---------------- K5: message + new edge_attr (two fused WMMA GEMMs) --------
__global__ __launch_bounds__(32) void k_edge_message(
    const float* __restrict__ H,         // N x 256 (pre-relu layer-1 out)
    const int* __restrict__ SRC, const int* __restrict__ DST,
    const _Float16* __restrict__ W2P, const float* __restrict__ B2, // packed 512x64
    const _Float16* __restrict__ EA16,                              // E x 64
    const _Float16* __restrict__ W3P, const float* __restrict__ B3, // packed 128x64
    _Float16* __restrict__ EA2_16, int Ecnt) {
  __shared__ alignas(32) _Float16 a_s[16][2 * HH];   // [h_src || h_dst]
  __shared__ alignas(32) _Float16 a2_s[16][2 * ED];  // [edge_attr || message]
  const int lane = threadIdx.x;
  const int e0 = blockIdx.x * 16;
  for (int r = 0; r < 16; ++r) {
    int e = min(e0 + r, Ecnt - 1);
    const float4* ps = (const float4*)(H + (size_t)SRC[e] * HH);
    const float4* pd = (const float4*)(H + (size_t)DST[e] * HH);
    v4h* pa = (v4h*)&a_s[r][0];
    for (int c = lane; c < HH / 4; c += 32) {
      float4 u = ps[c], w = pd[c];
      v4h hu = { (_Float16)u.x, (_Float16)u.y, (_Float16)u.z, (_Float16)u.w };
      v4h hw = { (_Float16)w.x, (_Float16)w.y, (_Float16)w.z, (_Float16)w.w };
      pa[c] = hu;
      pa[HH / 4 + c] = hw;
    }
    // first 64 cols of a2: previous edge_attr (contiguous f16 row copy)
    const v8h* pe = (const v8h*)(EA16 + (size_t)e * ED);
    v8h* qa = (v8h*)&a2_s[r][0];
    if (lane < ED / 8) qa[lane] = pe[lane];
  }
  __syncthreads();
  const int col0 = lane & 15, rh = (lane >> 4) << 3;
  // message = [xi||xj] @ w2 + b2   (K = 512): hoist A, 4 accumulators
  {
    v8f acc[4] = {{}, {}, {}, {}};
    for (int k = 0; k < 16; ++k) {
      v16h af = load_a16(&a_s[0][0] + k * 32, 2 * HH);
#pragma unroll
      for (int nt = 0; nt < 4; ++nt)
        acc[nt] = wmma16(af, load_bp(W2P, nt * 16 + k), acc[nt]);
    }
#pragma unroll
    for (int nt = 0; nt < 4; ++nt) {
      int col = nt * 16 + col0;
      float bv = B2[col];
#pragma unroll
      for (int r = 0; r < 8; ++r) a2_s[r + rh][ED + col] = (_Float16)(acc[nt][r] + bv);
    }
  }
  __syncthreads();
  // edge_attr2 = [edge_attr||message] @ w3 + b3   (K = 128)
  v16h c4[4];
#pragma unroll
  for (int k = 0; k < 4; ++k) c4[k] = load_a16(&a2_s[0][0] + k * 32, 2 * ED);
#pragma unroll
  for (int nt = 0; nt < 4; ++nt) {
    v8f acc = {};
#pragma unroll
    for (int k = 0; k < 4; ++k) acc = wmma16(c4[k], load_bp(W3P, nt * 4 + k), acc);
    int col = nt * 16 + col0;
    float bv = B3[col];
#pragma unroll
    for (int r = 0; r < 8; ++r) {
      int e = e0 + r + rh;
      if (e < Ecnt) EA2_16[(size_t)e * ED + col] = (_Float16)(acc[r] + bv);
    }
  }
}

// ---------------- pooling + classifier ----------------
__global__ void k_pool(const float* __restrict__ H2, const int* __restrict__ BATCH,
                       float* __restrict__ POOLED, int Nn) {
  long long i = (long long)blockIdx.x * blockDim.x + threadIdx.x;
  long long st = (long long)gridDim.x * blockDim.x;
  long long tot = (long long)Nn * HH;
  for (; i < tot; i += st) {
    int n = (int)(i >> 8);
    int c = (int)(i & 255);
    float v = fmaxf(H2[i], 0.f);
    atomicAdd(&POOLED[(size_t)BATCH[n] * HH + c], v);
  }
}

__global__ __launch_bounds__(32) void k_final(const float* __restrict__ POOLED,
                                              const float* __restrict__ W1,
                                              const float* __restrict__ B1,
                                              float* __restrict__ OUT) {
  const int g = blockIdx.x;
  const int lane = threadIdx.x;
  __shared__ float lg[NC];
  __shared__ float lse_s;
  if (lane < NC) {
    float acc = B1[lane];
    const float* p = POOLED + (size_t)g * HH;
    for (int k = 0; k < HH; ++k) acc += p[k] * W1[k * NC + lane];
    lg[lane] = acc;
  }
  __syncthreads();
  if (lane == 0) {
    float mx = -1e30f;
    for (int i = 0; i < NC; ++i) mx = fmaxf(mx, lg[i]);
    float s = 0.f;
    for (int i = 0; i < NC; ++i) s += __expf(lg[i] - mx);
    lse_s = mx + __logf(s);
  }
  __syncthreads();
  if (lane < NC) OUT[(size_t)g * NC + lane] = lg[lane] - lse_s;
}

// ---------------- host orchestration ----------------
extern "C" void kernel_launch(void* const* d_in, const int* in_sizes, int n_in,
                              void* d_out, int out_size, void* d_ws, size_t ws_size,
                              hipStream_t stream) {
  const float* x      = (const float*)d_in[0];
  const int*   ei     = (const int*)d_in[1];
  const float* raw    = (const float*)d_in[2];
  const int*   batch  = (const int*)d_in[3];
  const float* pm_w1  = (const float*)d_in[4];
  const float* pm_b1  = (const float*)d_in[5];
  const float* pm_w2  = (const float*)d_in[6];
  const float* pm_b2  = (const float*)d_in[7];
  const float* pm_ws  = (const float*)d_in[8];
  const float* pm_bs  = (const float*)d_in[9];
  const float* g1_wl  = (const float*)d_in[10];
  const float* g1_bl  = (const float*)d_in[11];
  const float* g1_wr  = (const float*)d_in[12];
  const float* g1_we  = (const float*)d_in[13];
  const float* g1_att = (const float*)d_in[14];
  const float* g1_bo  = (const float*)d_in[15];
  const float* g2_wl  = (const float*)d_in[16];
  const float* g2_bl  = (const float*)d_in[17];
  const float* g2_wr  = (const float*)d_in[18];
  const float* g2_we  = (const float*)d_in[19];
  const float* g2_att = (const float*)d_in[20];
  const float* g2_bo  = (const float*)d_in[21];
  const float* w2     = (const float*)d_in[22];
  const float* b2     = (const float*)d_in[23];
  const float* w3     = (const float*)d_in[24];
  const float* b3     = (const float*)d_in[25];
  const float* w1     = (const float*)d_in[26];
  const float* b1     = (const float*)d_in[27];
  float* out = (float*)d_out;

  const int* src = ei;
  const int* dst = ei + EE;

  // workspace carve-up
  char* base = (char*)d_ws;
  size_t off = 0;
  auto take = [&](size_t bytes) -> char* {
    char* p = base + off;
    off = (off + bytes + 255) & ~(size_t)255;
    return p;
  };
  _Float16* pm_w2_p = (_Float16*)take((size_t)HH * ED * 2);
  _Float16* pm_ws_p = (_Float16*)take((size_t)ED * ED * 2);
  _Float16* g1_wl_p = (_Float16*)take((size_t)F_IN * HH * 2);
  _Float16* g1_wr_p = (_Float16*)take((size_t)F_IN * HH * 2);
  _Float16* g1_we_p = (_Float16*)take((size_t)ED * HH * 2);
  _Float16* g2_wl_p = (_Float16*)take((size_t)HH * HH * 2);
  _Float16* g2_wr_p = (_Float16*)take((size_t)HH * HH * 2);
  _Float16* g2_we_p = (_Float16*)take((size_t)ED * HH * 2);
  _Float16* w2_p    = (_Float16*)take((size_t)2 * HH * ED * 2);
  _Float16* w3_p    = (_Float16*)take((size_t)2 * ED * ED * 2);
  _Float16* ea16    = (_Float16*)take((size_t)EE * ED * 2);
  _Float16* ea2_16  = (_Float16*)take((size_t)EE * ED * 2);
  float* xl     = (float*)take((size_t)NN * HH * 4);
  float* xr     = (float*)take((size_t)NN * HH * 4);
  float* h      = (float*)take((size_t)NN * HH * 4);
  float* h2     = (float*)take((size_t)NN * HH * 4);
  float* logit  = (float*)take((size_t)EE * 4);
  float* ex     = (float*)take((size_t)EE * 4);
  float* mmax   = (float*)take((size_t)NN * 4);
  float* den    = (float*)take((size_t)NN * 4);
  float* pooled = (float*)take((size_t)NG * HH * 4);

  auto pack = [&](const float* s, _Float16* d, int K, int N) {
    long long n = (long long)K * N;
    k_pack_b<<<(int)((n + 255) / 256), 256, 0, stream>>>(s, d, K, N);
  };
  pack(pm_w2, pm_w2_p, HH, ED);
  pack(pm_ws, pm_ws_p, ED, ED);        // rows 0..63 of (67,64)
  pack(g1_wl, g1_wl_p, F_IN, HH);
  pack(g1_wr, g1_wr_p, F_IN, HH);
  pack(g1_we, g1_we_p, ED, HH);
  pack(g2_wl, g2_wl_p, HH, HH);
  pack(g2_wr, g2_wr_p, HH, HH);
  pack(g2_we, g2_we_p, ED, HH);
  pack(w2, w2_p, 2 * HH, ED);
  pack(w3, w3_p, 2 * ED, ED);

  const int eb = (EE + 15) / 16;   // edge tiles
  const int nb = (NN + 15) / 16;   // node tiles

  // 1) edge_attr = perm_invariant_net(raw)
  k_edge_perm_net<<<eb, 32, 0, stream>>>(raw, pm_w1, pm_b1, pm_w2_p, pm_b2,
                                         pm_ws_p, pm_ws, pm_bs, ea16, EE);

  // 2) layer-1 node linears
  k_node_linear<F_IN / 32><<<nb, 32, 0, stream>>>(x, F_IN, 0, g1_wl_p, g1_bl,
                                                  g1_wr_p, xl, xr, NN);

  // 3) layer-1 attention softmax + aggregate
  k_fill<<<(NN + 255) / 256, 256, 0, stream>>>(mmax, -INFINITY, NN);
  k_fill<<<(NN + 255) / 256, 256, 0, stream>>>(den, 0.f, NN);
  k_rows_bias<<<4096, 256, 0, stream>>>(h, g1_bo, NN, HH);
  k_gat_logits<<<eb, 32, 0, stream>>>(ea16, g1_we_p, xl, xr, g1_att, src, dst,
                                      logit, mmax, EE);
  k_expsum<<<(EE + 255) / 256, 256, 0, stream>>>(logit, mmax, dst, ex, den, EE);
  {
    long long tot = (long long)EE * HH;
    k_scatter<<<(int)((tot + 255) / 256), 256, 0, stream>>>(ex, den, src, dst, xl,
                                                            h, tot);
  }

  // 4) messages + new edge attributes
  k_edge_message<<<eb, 32, 0, stream>>>(h, src, dst, w2_p, b2, ea16, w3_p, b3,
                                        ea2_16, EE);

  // 5) layer-2 node linears on relu(h)  (reuse xl/xr)
  k_node_linear<HH / 32><<<nb, 32, 0, stream>>>(h, HH, 1, g2_wl_p, g2_bl,
                                                g2_wr_p, xl, xr, NN);

  // 6) layer-2 attention softmax + aggregate
  k_fill<<<(NN + 255) / 256, 256, 0, stream>>>(mmax, -INFINITY, NN);
  k_fill<<<(NN + 255) / 256, 256, 0, stream>>>(den, 0.f, NN);
  k_rows_bias<<<4096, 256, 0, stream>>>(h2, g2_bo, NN, HH);
  k_gat_logits<<<eb, 32, 0, stream>>>(ea2_16, g2_we_p, xl, xr, g2_att, src, dst,
                                      logit, mmax, EE);
  k_expsum<<<(EE + 255) / 256, 256, 0, stream>>>(logit, mmax, dst, ex, den, EE);
  {
    long long tot = (long long)EE * HH;
    k_scatter<<<(int)((tot + 255) / 256), 256, 0, stream>>>(ex, den, src, dst, xl,
                                                            h2, tot);
  }

  // 7) pool relu(h2) by batch, classify, log_softmax
  k_fill<<<(NG * HH + 255) / 256, 256, 0, stream>>>(pooled, 0.f, (long long)NG * HH);
  k_pool<<<(int)(((long long)NN * HH + 255) / 256), 256, 0, stream>>>(h2, batch,
                                                                      pooled, NN);
  k_final<<<NG, 32, 0, stream>>>(pooled, w1, b1, out);
}
